// EfficientAdditiveAttnetion_65180423685605
// MI455X (gfx1250) — compile-verified, matrix operand order checked by reading, not run on hardware
//
#include <hip/hip_runtime.h>

typedef __attribute__((ext_vector_type(16))) __bf16 v16bf;
typedef __attribute__((ext_vector_type(8)))  __bf16 v8bf;
typedef __attribute__((ext_vector_type(4)))  __bf16 v4bf;
typedef __attribute__((ext_vector_type(8)))  float  v8f;

#define B_   16
#define N_   4096
#define DIN  512
#define HD   512
#define TD   256
#define EPS_ 1e-12f
#define SCALE_ 0.04419417382415922f   // 512^-0.5

#define BM 128
#define BN 64
#define BK 32
#define APAD 8   // row stride BK+8 elems = 80 bytes: 16B-aligned, conflict-free

// ---------------------------------------------------------------------------
// Generic bf16-WMMA GEMM:  C[M,Nout] = A[M,K] (scaled) @ W[K,Nout] + bias (+add)
// Software-pipelined, double-buffered LDS. Compile-time specialized:
//   ABF16 (A is bf16), RSCALE (per-row scale), CSCALE (per-(batch,k) scale,
//   batch = m>>batchShift), ADDB (bf16 residual), OUTF (f32 out, else bf16).
// 256 threads = 8 waves; wave w computes rows [w*16, w*16+16) x 64 cols.
// ---------------------------------------------------------------------------
template <bool ABF16, bool RSCALE, bool CSCALE, bool ADDB, bool OUTF>
__global__ __launch_bounds__(256) void wmma_gemm(
    const float*  __restrict__ Af, const __bf16* __restrict__ Ab,
    const float*  __restrict__ W,  const float*  __restrict__ bias,
    const float*  __restrict__ rowScale, const float* __restrict__ colScale,
    int batchShift,
    const __bf16* __restrict__ addB,
    float* __restrict__ outF, __bf16* __restrict__ outB,
    int M, int K, int Nout)
{
    __shared__ __bf16 As [2][BM][BK + APAD];   // row-major A tiles (ping-pong)
    __shared__ __bf16 Bst[2][BN][BK + APAD];   // TRANSPOSED W tiles: Bst[.][col][k]

    const int m0   = blockIdx.y * BM;
    const int n0   = blockIdx.x * BN;
    const int t    = threadIdx.x;
    const int wave = t >> 5;
    const int lane = t & 31;

    // register staging for one K-tile
    float4 aF[4];  v4bf aB4[4];  float rsReg[4];  float4 csReg[4];  float4 wR[2];

    auto loadTile = [&](int k0) {
#pragma unroll
        for (int i = 0; i < 4; ++i) {               // A: 128x32 as float4/v4bf units
            int u  = t + i * 256;
            int r  = u >> 3;                        // BK/4 = 8 units per row
            int c  = (u & 7) * 4;
            int gm = m0 + r, gk = k0 + c;
            if constexpr (ABF16) aB4[i] = *(const v4bf*)(Ab + (size_t)gm * K + gk);
            else                 aF[i]  = *(const float4*)(Af + (size_t)gm * K + gk);
            if constexpr (RSCALE) rsReg[i] = rowScale[gm];
            if constexpr (CSCALE)
                csReg[i] = *(const float4*)(colScale + (size_t)(gm >> batchShift) * K + gk);
            // branchless prefetch of the following K-tile (uniform clamped offset)
            int po = (k0 + BK < K) ? BK : 0;
            if constexpr (ABF16) __builtin_prefetch(Ab + (size_t)gm * K + gk + po, 0, 3);
            else                 __builtin_prefetch(Af + (size_t)gm * K + gk + po, 0, 3);
        }
#pragma unroll
        for (int i = 0; i < 2; ++i) {               // W: 32x64 as float4 units
            int u = t + i * 256;
            int r = u >> 4;                         // BN/4 = 16 units per row
            int c = (u & 15) * 4;
            wR[i] = *(const float4*)(W + (size_t)(k0 + r) * Nout + n0 + c);
        }
    };

    auto storeTile = [&](int buf) {
#pragma unroll
        for (int i = 0; i < 4; ++i) {
            int u = t + i * 256;
            int r = u >> 3;
            int c = (u & 7) * 4;
            v4bf o;
            if constexpr (ABF16 && !RSCALE && !CSCALE) {
                o = aB4[i];
            } else {
                float f0, f1, f2, f3;
                if constexpr (ABF16) {
                    f0 = (float)aB4[i][0]; f1 = (float)aB4[i][1];
                    f2 = (float)aB4[i][2]; f3 = (float)aB4[i][3];
                } else {
                    f0 = aF[i].x; f1 = aF[i].y; f2 = aF[i].z; f3 = aF[i].w;
                }
                if constexpr (RSCALE) {
                    f0 *= rsReg[i]; f1 *= rsReg[i]; f2 *= rsReg[i]; f3 *= rsReg[i];
                }
                if constexpr (CSCALE) {
                    f0 *= csReg[i].x; f1 *= csReg[i].y; f2 *= csReg[i].z; f3 *= csReg[i].w;
                }
                o[0] = (__bf16)f0; o[1] = (__bf16)f1; o[2] = (__bf16)f2; o[3] = (__bf16)f3;
            }
            *(v4bf*)&As[buf][r][c] = o;
        }
#pragma unroll
        for (int i = 0; i < 2; ++i) {
            int u = t + i * 256;
            int r = u >> 4;
            int c = (u & 15) * 4;
            Bst[buf][c + 0][r] = (__bf16)wR[i].x;
            Bst[buf][c + 1][r] = (__bf16)wR[i].y;
            Bst[buf][c + 2][r] = (__bf16)wR[i].z;
            Bst[buf][c + 3][r] = (__bf16)wR[i].w;
        }
    };

    v8f acc[4];
#pragma unroll
    for (int f = 0; f < 4; ++f) acc[f] = (v8f){0,0,0,0,0,0,0,0};

    const int nt = K / BK;
    loadTile(0);
    storeTile(0);
    __syncthreads();

    const int arow = wave * 16 + (lane & 15);
    const int kh   = (lane >> 4) * 8;

    for (int kt = 0; kt < nt; ++kt) {
        const int cur = kt & 1;
        if (kt + 1 < nt) loadTile((kt + 1) * BK);   // global loads fly under WMMAs

        // ---- A fragment: two contiguous ds_load_b128 per lane (CDNA5 layout)
        v8bf alo = *(const v8bf*)&As[cur][arow][kh];
        v8bf ahi = *(const v8bf*)&As[cur][arow][kh + 16];
        v16bf a;
#pragma unroll
        for (int j = 0; j < 8; ++j) { a[j] = alo[j]; a[8 + j] = ahi[j]; }

        // ---- 4 B fragments (same contiguous pattern on transposed tile) + WMMA
#pragma unroll
        for (int f = 0; f < 4; ++f) {
            const int bcol = f * 16 + (lane & 15);
            v8bf blo = *(const v8bf*)&Bst[cur][bcol][kh];
            v8bf bhi = *(const v8bf*)&Bst[cur][bcol][kh + 16];
            v16bf b;
#pragma unroll
            for (int j = 0; j < 8; ++j) { b[j] = blo[j]; b[8 + j] = bhi[j]; }
            acc[f] = __builtin_amdgcn_wmma_f32_16x16x32_bf16(
                false, a, false, b, (short)0, acc[f], false, false);
        }

        if (kt + 1 < nt) storeTile(1 - cur);        // fill the other buffer
        __syncthreads();                            // single barrier per K-step
    }

    // ---- epilogue: C layout row = v + 8*(lane>=16), col = lane&15
    const int col = lane & 15;
    const int rb  = (lane >> 4) * 8;
#pragma unroll
    for (int f = 0; f < 4; ++f) {
        const int gn   = n0 + f * 16 + col;
        const float bv = bias[gn];
#pragma unroll
        for (int v = 0; v < 8; ++v) {
            int gm = m0 + wave * 16 + rb + v;
            float val = acc[f][v] + bv;
            if constexpr (ADDB) val += (float)addB[(size_t)gm * Nout + gn];
            if constexpr (OUTF) outF[(size_t)gm * Nout + gn] = val;
            else                outB[(size_t)gm * Nout + gn] = (__bf16)val;
        }
    }
}

// ---------------------------------------------------------------------------
// Per-row stats: kn_inv[m] = 1/max(||k_row||,eps)
//                qw[m]    = (q_row . w_g) / max(||q_row||,eps) * SCALE
// One wave32 per row (16 contiguous bf16 per lane, wide loads), 8 rows/block.
// ---------------------------------------------------------------------------
__global__ __launch_bounds__(256) void rowstats(
    const __bf16* __restrict__ Q, const __bf16* __restrict__ Kb,
    const float* __restrict__ wg,
    float* __restrict__ knInv, float* __restrict__ qw)
{
    const int row  = blockIdx.x * 8 + (threadIdx.x >> 5);
    const int lane = threadIdx.x & 31;
    const v8bf* q8 = (const v8bf*)(Q  + (size_t)row * HD);
    const v8bf* k8 = (const v8bf*)(Kb + (size_t)row * HD);
    float sq = 0.f, sk = 0.f, dq = 0.f;
#pragma unroll
    for (int u = 0; u < 2; ++u) {
        v8bf qv = q8[lane * 2 + u];
        v8bf kv = k8[lane * 2 + u];
        float4 w0 = *(const float4*)(wg + lane * 16 + u * 8);
        float4 w1 = *(const float4*)(wg + lane * 16 + u * 8 + 4);
        float wv[8] = {w0.x, w0.y, w0.z, w0.w, w1.x, w1.y, w1.z, w1.w};
#pragma unroll
        for (int e = 0; e < 8; ++e) {
            float q = (float)qv[e], k = (float)kv[e];
            sq += q * q; sk += k * k; dq += q * wv[e];
        }
    }
    for (int off = 16; off > 0; off >>= 1) {
        sq += __shfl_xor(sq, off);
        sk += __shfl_xor(sk, off);
        dq += __shfl_xor(dq, off);
    }
    if (lane == 0) {
        knInv[row] = 1.f / fmaxf(sqrtf(sk), EPS_);
        float qwv  = (dq / fmaxf(sqrtf(sq), EPS_)) * SCALE_;
        qw[row]    = qwv;
    }
}

// rinv[b] = 1/max(||qw[b,:]||, eps)  (deterministic block-tree reduction)
__global__ __launch_bounds__(256) void batch_rinv(
    const float* __restrict__ qw, float* __restrict__ rinv)
{
    __shared__ float s[256];
    const int b = blockIdx.x;
    float acc = 0.f;
    for (int n = (int)threadIdx.x; n < N_; n += 256) {
        float v = qw[b * N_ + n];
        acc += v * v;
    }
    s[threadIdx.x] = acc;
    __syncthreads();
    for (int st = 128; st > 0; st >>= 1) {
        if ((int)threadIdx.x < st) s[threadIdx.x] += s[threadIdx.x + st];
        __syncthreads();
    }
    if (threadIdx.x == 0) rinv[b] = 1.f / fmaxf(sqrtf(s[0]), EPS_);
}

// ---------------------------------------------------------------------------
// G[b,h] = rinv[b] * sum_n qw[b,n] * Q[b,n,h]  — two deterministic stages
// ---------------------------------------------------------------------------
#define GP_CHUNK 32
__global__ __launch_bounds__(256) void gpool_partial(
    const __bf16* __restrict__ Q, const float* __restrict__ qw,
    float* __restrict__ Gpart)
{
    const int h = blockIdx.x * 256 + threadIdx.x;
    const int b = blockIdx.y;
    const int c = blockIdx.z;
    const int nBeg = c * (N_ / GP_CHUNK);
    float acc = 0.f;
    const size_t base = ((size_t)b * N_ + nBeg) * HD;
    for (int n = 0; n < N_ / GP_CHUNK; ++n)
        acc += qw[b * N_ + nBeg + n] * (float)Q[base + (size_t)n * HD + h];
    Gpart[((size_t)c * B_ + b) * HD + h] = acc;
}

__global__ __launch_bounds__(256) void gpool_final(
    const float* __restrict__ Gpart, const float* __restrict__ rinv,
    float* __restrict__ G)
{
    const int h = blockIdx.x * 256 + threadIdx.x;
    const int b = blockIdx.y;
    float acc = 0.f;
    for (int c = 0; c < GP_CHUNK; ++c)
        acc += Gpart[((size_t)c * B_ + b) * HD + h];
    G[b * HD + h] = acc * rinv[b];
}

// ---------------------------------------------------------------------------
extern "C" void kernel_launch(void* const* d_in, const int* in_sizes, int n_in,
                              void* d_out, int out_size, void* d_ws, size_t ws_size,
                              hipStream_t stream)
{
    (void)in_sizes; (void)n_in; (void)out_size; (void)ws_size;
    const float* x   = (const float*)d_in[0];
    const float* Wq  = (const float*)d_in[1];
    const float* bq  = (const float*)d_in[2];
    const float* Wk  = (const float*)d_in[3];
    const float* bk  = (const float*)d_in[4];
    const float* wg  = (const float*)d_in[5];
    const float* Wp  = (const float*)d_in[6];
    const float* bp  = (const float*)d_in[7];
    const float* Wf  = (const float*)d_in[8];
    const float* bfb = (const float*)d_in[9];

    const size_t M = (size_t)B_ * N_;
    char* ws = (char*)d_ws;
    __bf16* Qbf  = (__bf16*)ws;  ws += M * HD * 2;
    __bf16* Kbf  = (__bf16*)ws;  ws += M * HD * 2;
    __bf16* O1bf = (__bf16*)ws;  ws += M * HD * 2;
    float* knInv = (float*)ws;   ws += M * 4;
    float* qw    = (float*)ws;   ws += M * 4;
    float* rinv  = (float*)ws;   ws += 256;
    float* G     = (float*)ws;   ws += (size_t)B_ * HD * 4;
    float* Gpart = (float*)ws;   ws += (size_t)GP_CHUNK * B_ * HD * 4;

    dim3 blk(256);
    dim3 gQK(HD / BN, (unsigned)(M / BM));

    // Q = x@Wq + bq  (bf16); K = x@Wk + bk (bf16)
    wmma_gemm<false,false,false,false,false><<<gQK, blk, 0, stream>>>(
        x, nullptr, Wq, bq, nullptr, nullptr, 0,
        nullptr, nullptr, Qbf, (int)M, DIN, HD);
    wmma_gemm<false,false,false,false,false><<<gQK, blk, 0, stream>>>(
        x, nullptr, Wk, bk, nullptr, nullptr, 0,
        nullptr, nullptr, Kbf, (int)M, DIN, HD);

    // row norms + w_g dot
    rowstats<<<dim3((unsigned)(M / 8)), blk, 0, stream>>>(Qbf, Kbf, wg, knInv, qw);
    // per-batch sequence-dim normalization
    batch_rinv<<<dim3(B_), blk, 0, stream>>>(qw, rinv);
    // weighted pool G = sum_n A*q
    gpool_partial<<<dim3(HD / 256, B_, GP_CHUNK), blk, 0, stream>>>(Qbf, qw, Gpart);
    gpool_final<<<dim3(HD / 256, B_), blk, 0, stream>>>(Gpart, rinv, G);

    // out1 = (G * key_norm) @ Wp + bp + query   (bf16)
    wmma_gemm<true,true,true,true,false><<<gQK, blk, 0, stream>>>(
        nullptr, Kbf, Wp, bp, knInv, G, 12,
        Qbf, nullptr, O1bf, (int)M, HD, HD);

    // final = out1 @ Wf + bf   (f32 -> d_out)
    dim3 gF(TD / BN, (unsigned)(M / BM));
    wmma_gemm<true,false,false,false,true><<<gF, blk, 0, stream>>>(
        nullptr, O1bf, Wf, bfb, nullptr, nullptr, 0,
        nullptr, (float*)d_out, nullptr, (int)M, HD, TD);
}
